// BehaviorTier_37821482009254
// MI455X (gfx1250) — matrix operand order repaired and checked by reading, not compiled, not run on hardware
//
#include <hip/hip_runtime.h>

typedef _Float16 f16;
typedef __attribute__((ext_vector_type(16))) _Float16 v16h;
typedef __attribute__((ext_vector_type(8)))  _Float16 v8h;
typedef __attribute__((ext_vector_type(8)))  float    v8f;

#define DHID 64
#define SEQ  256
#define NW   2          // waves per block
#define WSTR 72         // padded LDS row stride (halfs) for weights
#define HSTR 72         // padded LDS row stride (halfs) for activation staging

__device__ __forceinline__ v16h cat8(v8h lo, v8h hi) {
  v16h r;
  #pragma unroll
  for (int i = 0; i < 8; ++i) { r[i] = lo[i]; r[8 + i] = hi[i]; }
  return r;
}

// Launder a 32-bit offset through an empty asm so LICM cannot hoist the
// (actually loop-invariant) LDS weight-fragment loads.  Keeping the base as a
// known-LDS pointer lets the backend still emit ds_load_b128 (not flat_load).
__device__ __forceinline__ int launder_i(int v) {
  asm volatile("" : "+v"(v));
  return v;
}

// Branch-free tanh: native v_tanh_f32 if the toolchain exposes it, otherwise
// tanh(x) = 1 - 2/(exp2(2*log2e*x)+1) via v_exp_f32 + v_rcp_f32.
// Saturates to +/-1 for large |x|; no EXEC-mask branching (WMMA-friendly).
__device__ __forceinline__ float fast_tanh(float x) {
#if __has_builtin(__builtin_amdgcn_tanhf)
  return __builtin_amdgcn_tanhf(x);
#elif __has_builtin(__builtin_amdgcn_tanh_f32)
  return __builtin_amdgcn_tanh_f32(x);
#else
  const float t = __builtin_amdgcn_exp2f(x * 2.8853900817779268f); // 2*log2(e)
  return 1.0f - 2.0f * __builtin_amdgcn_rcpf(t + 1.0f);
#endif
}

__device__ __forceinline__ v8f tanh8(v8f v) {
  v8f r;
  #pragma unroll
  for (int j = 0; j < 8; ++j) r[j] = fast_tanh(v[j]);
  return r;
}

// A fragment (16x32 f16, ktile selects K[0,32) or K[32,64)) from LDS tile [16][HSTR].
// Lanes 0-15 hold M=lane, K = {k0..k0+7, k0+16..k0+23}, k0 = 0 (8 for lanes 16-31).
__device__ __forceinline__ v16h load_A_lds(const f16* sh, int lane, int ktile) {
  const int m  = lane & 15;
  const int k0 = ktile * 32 + ((lane < 16) ? 0 : 8);
  v8h lo = *(const v8h*)(sh + m * HSTR + k0);
  v8h hi = *(const v8h*)(sh + m * HSTR + k0 + 16);
  return cat8(lo, hi);
}

// A fragment of the x_t tile straight from global f32 (xrow = &x[bbase+m][s][0]).
__device__ __forceinline__ v16h load_A_gx(const float* xrow, int lane, int ktile) {
  const int k0 = ktile * 32 + ((lane < 16) ? 0 : 8);
  v16h a;
  #pragma unroll
  for (int i = 0; i < 8; ++i) {
    a[i]     = (f16)xrow[k0 + i];
    a[8 + i] = (f16)xrow[k0 + 16 + i];
  }
  return a;
}

// B fragment (32x16 f16): B[k][n] = W[n][k]; sw = LDS weights row-major [64][WSTR].
// Lanes 0-15: n=lane, K = {kb..kb+7, kb+16..kb+23} with kb = ktile*32; lanes 16-31: same n, kb += 8.
__device__ __forceinline__ v16h load_B_lds(const f16* sw, int lane, int ktile, int ntile) {
  const int n  = (lane & 15) + ntile * 16;
  const int kb = ktile * 32 + ((lane < 16) ? 0 : 8);
  v8h lo = *(const v8h*)(sw + n * WSTR + kb);
  v8h hi = *(const v8h*)(sw + n * WSTR + kb + 16);
  return cat8(lo, hi);
}

__device__ __forceinline__ v8f wmma_f16(v16h a, v16h b, v8f c) {
  return __builtin_amdgcn_wmma_f32_16x16x32_f16(false, a, false, b, (short)0, c, false, false);
}

// Store four 16x16 f32 C fragments into the wave's LDS tile as f16, row-major [16][HSTR].
// C layout: lane holds column n = lane&15 (+ntile*16), rows m = j + (lane<16 ? 0 : 8).
__device__ __forceinline__ void store_C_lds(f16* sh, int lane, const v8f c[4]) {
  const int n  = lane & 15;
  const int mb = (lane < 16) ? 0 : 8;
  #pragma unroll
  for (int nt = 0; nt < 4; ++nt)
    #pragma unroll
    for (int j = 0; j < 8; ++j)
      sh[(mb + j) * HSTR + nt * 16 + n] = (f16)c[nt][j];
}

// One (16,64)x(64,64) GEMM: 8 WMMAs + per-column bias.
__device__ __forceinline__ void gemm4(v16h a0, v16h a1, const f16* sw, int lane,
                                      const float biasrow[4], v8f c[4]) {
  #pragma unroll
  for (int nt = 0; nt < 4; ++nt) {
    v8f acc = {};
    acc = wmma_f16(a0, load_B_lds(sw, lane, 0, nt), acc);
    acc = wmma_f16(a1, load_B_lds(sw, lane, 1, nt), acc);
    #pragma unroll
    for (int j = 0; j < 8; ++j) acc[j] += biasrow[nt];
    c[nt] = acc;
  }
}

__global__ __launch_bounds__(32 * NW)
void rnn_scan_kernel(const float* __restrict__ x,   const float* __restrict__ pe,
                     const float* __restrict__ Wnh, const float* __restrict__ Wh,
                     const float* __restrict__ Wnr, const float* __restrict__ Wnt,
                     const float* __restrict__ Wb,
                     const float* __restrict__ bnh, const float* __restrict__ bh,
                     const float* __restrict__ bnr, const float* __restrict__ bnt,
                     const float* __restrict__ bb,
                     float* __restrict__ out) {
  __shared__ f16 sW[5 * 64 * WSTR];     // 5 weight matrices, f16, padded rows
  __shared__ f16 sH[NW * 16 * HSTR];    // per-wave activation staging tile

  const int tid  = threadIdx.x;
  const int wave = tid >> 5;
  const int lane = tid & 31;

  // Cooperative weight staging: f32 global -> f16 LDS, W stored [n][k] row-major.
  const float* wsrc[5] = {Wnh, Wh, Wnr, Wnt, Wb};
  #pragma unroll
  for (int w = 0; w < 5; ++w)
    for (int i = tid; i < 64 * 64; i += 32 * NW) {
      const int n = i >> 6, k = i & 63;
      sW[(w * 64 + n) * WSTR + k] = (f16)wsrc[w][i];
    }
  __syncthreads();

  f16* sh = sH + wave * 16 * HSTR;
  const int bbase = (blockIdx.x * NW + wave) * 16;
  const int n0 = lane & 15;
  const int mb = (lane < 16) ? 0 : 8;

  // Per-lane bias fragments (column n0 + 16*nt), loaded once.
  float bias[5][4];
  const float* bsrc[5] = {bnh, bh, bnr, bnt, bb};
  #pragma unroll
  for (int w = 0; w < 5; ++w)
    #pragma unroll
    for (int nt = 0; nt < 4; ++nt)
      bias[w][nt] = bsrc[w][nt * 16 + n0];

  // h0 = 0, staged as f16.
  for (int i = lane; i < 16 * 64; i += 32)
    sh[(i >> 6) * HSTR + (i & 63)] = (f16)0.f;

  const float* xrow = x  + (size_t)(bbase + n0) * SEQ * DHID;  // A-frag row base
  const float* prow = pe + (size_t)(bbase + mb) * SEQ * DHID;  // C-frag row base

  v8f hnext[4];
  for (int s = 0; s < SEQ; ++s) {
    const size_t so = (size_t)s * DHID;
    // Loop-variant (to the compiler) LDS offsets -> loads stay ds_load, not hoisted.
    const f16* w0 = sW + launder_i(0 * 64 * WSTR);
    const f16* w1 = sW + launder_i(1 * 64 * WSTR);
    const f16* w2 = sW + launder_i(2 * 64 * WSTR);
    const f16* w3 = sW + launder_i(3 * 64 * WSTR);
    const f16* w4 = sW + launder_i(4 * 64 * WSTR);

    v8f nh[4], hc[4], tmp[4];

    // nh = tanh(x_t @ Wnh^T + bnh)   (A-frags from global, convert f32->f16)
    {
      const float* xb = xrow + so;
      v16h a0 = load_A_gx(xb, lane, 0);
      v16h a1 = load_A_gx(xb, lane, 1);
      gemm4(a0, a1, w0, lane, bias[0], nh);
      #pragma unroll
      for (int nt = 0; nt < 4; ++nt) nh[nt] = tanh8(nh[nt]);
      if (s + 1 < SEQ) __builtin_prefetch(xb + DHID, 0, 0);
    }

    // h = tanh(h_prev @ Wh^T + bh); h_curr = h + nh
    {
      v16h a0 = load_A_lds(sh, lane, 0);
      v16h a1 = load_A_lds(sh, lane, 1);
      gemm4(a0, a1, w1, lane, bias[1], hc);
      #pragma unroll
      for (int nt = 0; nt < 4; ++nt) {
        v8f t = tanh8(hc[nt]);
        #pragma unroll
        for (int j = 0; j < 8; ++j) hc[nt][j] = t[j] + nh[nt][j];
      }
      store_C_lds(sh, lane, hc);   // stage h_curr
    }

    // nr = tanh(h_curr @ Wnr^T + bnr)
    {
      v16h a0 = load_A_lds(sh, lane, 0);
      v16h a1 = load_A_lds(sh, lane, 1);
      v8f nr[4];
      gemm4(a0, a1, w2, lane, bias[2], nr);
      #pragma unroll
      for (int nt = 0; nt < 4; ++nt) nr[nt] = tanh8(nr[nt]);
      store_C_lds(sh, lane, nr);   // stage nr
    }

    // nt = tanh(nr @ Wnt^T + bnt); h_next = h_curr + nt
    {
      v16h a0 = load_A_lds(sh, lane, 0);
      v16h a1 = load_A_lds(sh, lane, 1);
      gemm4(a0, a1, w3, lane, bias[3], hnext);
      #pragma unroll
      for (int nt = 0; nt < 4; ++nt) {
        v8f t = tanh8(hnext[nt]);
        #pragma unroll
        for (int j = 0; j < 8; ++j) hnext[nt][j] = hc[nt][j] + t[j];
      }
    }

    // tmp = h_next + p_t   (gather pos_enc in C-frag layout)
    #pragma unroll
    for (int nt = 0; nt < 4; ++nt)
      #pragma unroll
      for (int j = 0; j < 8; ++j)
        tmp[nt][j] = hnext[nt][j] + prow[(size_t)j * SEQ * DHID + so + nt * 16 + n0];
    store_C_lds(sh, lane, tmp);

    // h_next += tanh(tmp @ Wb^T + bb)
    {
      v16h a0 = load_A_lds(sh, lane, 0);
      v16h a1 = load_A_lds(sh, lane, 1);
      v8f bout[4];
      gemm4(a0, a1, w4, lane, bias[4], bout);
      #pragma unroll
      for (int nt = 0; nt < 4; ++nt) {
        v8f t = tanh8(bout[nt]);
        #pragma unroll
        for (int j = 0; j < 8; ++j) hnext[nt][j] += t[j];
      }
    }

    store_C_lds(sh, lane, hnext);  // re-stage h for the next step
  }

  // Write h_final (B, D) f32 from C fragments.
  #pragma unroll
  for (int nt = 0; nt < 4; ++nt)
    #pragma unroll
    for (int j = 0; j < 8; ++j)
      out[(size_t)(bbase + mb + j) * DHID + nt * 16 + n0] = hnext[nt][j];
}

extern "C" void kernel_launch(void* const* d_in, const int* in_sizes, int n_in,
                              void* d_out, int out_size, void* d_ws, size_t ws_size,
                              hipStream_t stream) {
  (void)in_sizes; (void)n_in; (void)d_ws; (void)ws_size; (void)out_size;
  const float* x   = (const float*)d_in[0];
  const float* pe  = (const float*)d_in[1];
  const float* Wnh = (const float*)d_in[2];
  const float* bnh = (const float*)d_in[3];
  const float* Wh  = (const float*)d_in[4];
  const float* bh  = (const float*)d_in[5];
  const float* Wnr = (const float*)d_in[6];
  const float* bnr = (const float*)d_in[7];
  const float* Wnt = (const float*)d_in[8];
  const float* bnt = (const float*)d_in[9];
  const float* Wb  = (const float*)d_in[10];
  const float* bb  = (const float*)d_in[11];
  float* out = (float*)d_out;

  dim3 grid(2048 / (16 * NW));   // 64 blocks, 16 batch rows per wave
  dim3 block(32 * NW);           // 2 waves per block
  hipLaunchKernelGGL(rnn_scan_kernel, grid, block, 0, stream,
                     x, pe, Wnh, Wh, Wnr, Wnt, Wb, bnh, bh, bnr, bnt, bb, out);
}
